// MaxAttention_36988258353456
// MI455X (gfx1250) — matrix operand order repaired and checked by reading, not compiled, not run on hardware
//
#include <hip/hip_runtime.h>
#include <hip/hip_bf16.h>

// ---- problem constants (from reference) ----
#define Bx 2
#define Nx 1024
#define DIMx 768
#define Rx 4
#define INNERx 768
#define Hx 12
#define DHx 64
#define SCALEx 0.125f   // 64^-0.5
#define WAVE 32

typedef __bf16 bf16_t;
typedef __attribute__((ext_vector_type(8)))  __bf16 v8bf;
typedef __attribute__((ext_vector_type(16))) __bf16 v16bf;
typedef __attribute__((ext_vector_type(8)))  float  v8f;

static __device__ __forceinline__ v16bf cat8(v8bf lo, v8bf hi) {
  return __builtin_shufflevector(lo, hi, 0,1,2,3,4,5,6,7,8,9,10,11,12,13,14,15);
}

// A fragment: 16(M) x 32(K) bf16, source row-major with row stride ld (elements).
// ISA layout: lane%16 = row; lanes<16 hold K 0..7 & 16..23; lanes>=16 hold K 8..15 & 24..31.
static __device__ __forceinline__ v16bf load_a_frag(const bf16_t* src, int ld, int lane) {
  int row  = lane & 15;
  int koff = (lane >> 4) * 8;
  const bf16_t* p = src + (size_t)row * ld + koff;
  v8bf lo = *(const v8bf*)(p);
  v8bf hi = *(const v8bf*)(p + 16);
  return cat8(lo, hi);
}

// B fragment: 32(K) x 16(N) bf16; element [k][n] lives at src[n*ld + k] (K contiguous).
// ISA layout: lane%16 = n; lanes<16 hold K 0..7 & 16..23; lanes>=16 hold K 8..15 & 24..31.
static __device__ __forceinline__ v16bf load_b_frag(const bf16_t* src, int ld, int lane) {
  int n  = lane & 15;
  int kg = (lane >> 4) * 8;
  const bf16_t* p = src + (size_t)n * ld + kg;
  v8bf lo = *(const v8bf*)(p);
  v8bf hi = *(const v8bf*)(p + 16);
  return cat8(lo, hi);
}

static __device__ __forceinline__ v8f wmma_bf16(v16bf a, v16bf b, v8f c) {
  return __builtin_amdgcn_wmma_f32_16x16x32_bf16(false, a, false, b, (short)0, c, false, false);
}

// ---------------- fp32 -> bf16 conversion ----------------
__global__ void k_cvt_bf16(const float* __restrict__ in, bf16_t* __restrict__ out, int n) {
  int i = blockIdx.x * blockDim.x + threadIdx.x;
  if (i < n) out[i] = (bf16_t)in[i];
}

// ---------------- QKV projection GEMM (64x32 per wave, 4x2 WMMA tiles) ----------------
// out[m, r, c] = sum_d x[m,d] * Wqkv[r,c,d]; q pre-scaled; v stored dh-major (transposed).
__global__ void __attribute__((amdgpu_waves_per_eu(1)))
k_qkv(const bf16_t* __restrict__ xb, const bf16_t* __restrict__ wq,
      bf16_t* __restrict__ qb, bf16_t* __restrict__ kb,
      bf16_t* __restrict__ vt) {
  const int CB_TOTAL = 72 + 3 * 48;   // 32-col blocks: route0 full 2304; routes 1..3 q,k only
  int wid  = blockIdx.x * (blockDim.x / WAVE) + (int)(threadIdx.x / WAVE);
  int lane = threadIdx.x & (WAVE - 1);
  if (wid >= 32 * CB_TOTAL) return;
  int rowBlk = wid / CB_TOTAL;
  int cb = wid % CB_TOTAL;
  int r, cblk;
  if (cb < 72) { r = 0; cblk = cb; }
  else { int t = cb - 72; r = 1 + t / 48; cblk = t % 48; }
  int mBase = rowBlk * 64;
  int cBase = cblk * 32;

  const bf16_t* A0 = xb + (size_t)mBase * DIMx;
  const bf16_t* B0 = wq + ((size_t)r * 3 * INNERx + cBase) * DIMx;

  v8f acc[4][2];
#pragma unroll
  for (int t = 0; t < 4; t++)
#pragma unroll
    for (int u = 0; u < 2; u++) acc[t][u] = (v8f){};

  for (int k0 = 0; k0 < DIMx; k0 += 32) {
    __builtin_prefetch(A0 + k0 + 128, 0, 1);
    __builtin_prefetch(B0 + k0 + 128, 0, 1);
    v16bf af[4], bf[2];
#pragma unroll
    for (int t = 0; t < 4; t++) af[t] = load_a_frag(A0 + (size_t)t * 16 * DIMx + k0, DIMx, lane);
#pragma unroll
    for (int u = 0; u < 2; u++) bf[u] = load_b_frag(B0 + (size_t)u * 16 * DIMx + k0, DIMx, lane);
#pragma unroll
    for (int t = 0; t < 4; t++)
#pragma unroll
      for (int u = 0; u < 2; u++) acc[t][u] = wmma_bf16(af[t], bf[u], acc[t][u]);
  }

  int half = lane >> 4;
#pragma unroll
  for (int t = 0; t < 4; t++)
#pragma unroll
    for (int u = 0; u < 2; u++) {
      int col = cBase + u * 16 + (lane & 15);
#pragma unroll
      for (int i = 0; i < 8; i++) {
        int m = mBase + t * 16 + i + 8 * half;
        int bb = m >> 10, nrow = m & (Nx - 1);
        float v = acc[t][u][i];
        if (col < INNERx) {                                   // q (pre-scaled)
          int h = col / DHx, d = col & (DHx - 1);
          qb[((((size_t)(bb * Hx + h)) * Rx + r) * Nx + nrow) * DHx + d] = (bf16_t)(v * SCALEx);
        } else if (col < 2 * INNERx) {                        // k
          int c2 = col - INNERx; int h = c2 / DHx, d = c2 & (DHx - 1);
          kb[((((size_t)(bb * Hx + h)) * Rx + r) * Nx + nrow) * DHx + d] = (bf16_t)v;
        } else if (r == 0) {                                  // v route 0, stored [b,h,dh,n]
          int c2 = col - 2 * INNERx; int h = c2 / DHx, d = c2 & (DHx - 1);
          vt[(((size_t)(bb * Hx + h)) * DHx + d) * Nx + nrow] = (bf16_t)v;
        }
      }
    }
}

// ---------------- attention pass 1: per-(route,row) softmax stats ----------------
// One wave handles 64 query rows (4 row-tiles); K-tile fragments reused 4x.
__global__ void __attribute__((amdgpu_waves_per_eu(1)))
k_attn_stats(const bf16_t* __restrict__ qb, const bf16_t* __restrict__ kb,
             float* __restrict__ mArr, float* __restrict__ lArr) {
  int wid  = blockIdx.x * (blockDim.x / WAVE) + (int)(threadIdx.x / WAVE);
  int lane = threadIdx.x & (WAVE - 1);
  if (wid >= Bx * Hx * Rx * (Nx / 64)) return;
  int rowBlk = wid & 15; int t0 = wid >> 4;
  int r = t0 % Rx; t0 /= Rx; int h = t0 % Hx; int b = t0 / Hx;

  size_t base = (((size_t)(b * Hx + h)) * Rx + r) * Nx * DHx;
  const bf16_t* Q  = qb + base + (size_t)rowBlk * 64 * DHx;
  const bf16_t* Kb = kb + base;

  v16bf aq[4][2];
#pragma unroll
  for (int t = 0; t < 4; t++) {
    aq[t][0] = load_a_frag(Q + (size_t)t * 16 * DHx, DHx, lane);
    aq[t][1] = load_a_frag(Q + (size_t)t * 16 * DHx + 32, DHx, lane);
  }

  float mrow[4][8], lrow[4][8];
#pragma unroll
  for (int t = 0; t < 4; t++)
#pragma unroll
    for (int i = 0; i < 8; i++) { mrow[t][i] = -1e30f; lrow[t][i] = 0.f; }

  for (int ct = 0; ct < Nx / 16; ct++) {
    const bf16_t* Bp = Kb + (size_t)ct * 16 * DHx;
    v16bf b0 = load_b_frag(Bp, DHx, lane);
    v16bf b1 = load_b_frag(Bp + 32, DHx, lane);
#pragma unroll
    for (int t = 0; t < 4; t++) {
      v8f s = {};
      s = wmma_bf16(aq[t][0], b0, s);
      s = wmma_bf16(aq[t][1], b1, s);
#pragma unroll
      for (int i = 0; i < 8; i++) {
        float v = s[i];
        float tm = v;
        for (int msk = 8; msk >= 1; msk >>= 1) tm = fmaxf(tm, __shfl_xor(tm, msk, WAVE));
        float nm = fmaxf(mrow[t][i], tm);
        float p = __expf(v - nm);
        for (int msk = 8; msk >= 1; msk >>= 1) p += __shfl_xor(p, msk, WAVE);
        lrow[t][i] = lrow[t][i] * __expf(mrow[t][i] - nm) + p;
        mrow[t][i] = nm;
      }
    }
  }
  if ((lane & 15) == 0) {
    size_t sb0 = ((((size_t)(b * Hx + h)) * Rx + r) * Nx) + (size_t)rowBlk * 64 + (lane >> 4) * 8;
#pragma unroll
    for (int t = 0; t < 4; t++)
#pragma unroll
      for (int i = 0; i < 8; i++) {
        mArr[sb0 + t * 16 + i] = mrow[t][i];
        lArr[sb0 + t * 16 + i] = lrow[t][i];
      }
  }
}

// ---------------- attention pass 2: p = exp(s-m)/l, max over routes, O = P.V ----------------
__global__ void __attribute__((amdgpu_waves_per_eu(1)))
k_attn_apply(const bf16_t* __restrict__ qb, const bf16_t* __restrict__ kb,
             const bf16_t* __restrict__ vt,
             const float* __restrict__ mArr, const float* __restrict__ lArr,
             bf16_t* __restrict__ ao) {
  __shared__ bf16_t lds[8][16][32];  // per-wave P tile staging (C-layout -> A-layout)
  int wslot = threadIdx.x / WAVE;
  int wid   = blockIdx.x * (blockDim.x / WAVE) + wslot;
  int lane  = threadIdx.x & (WAVE - 1);
  if (wid >= Bx * Hx * (Nx / 16)) return;
  int rowTile = wid & 63; int t = wid >> 6; int h = t % Hx; int b = t / Hx;
  int half = lane >> 4;

  float mv[Rx][8], lv[Rx][8];
#pragma unroll
  for (int r = 0; r < Rx; r++) {
    size_t sb = ((((size_t)(b * Hx + h)) * Rx + r) * Nx) + (size_t)rowTile * 16;
#pragma unroll
    for (int i = 0; i < 8; i++) {
      int row = i + 8 * half;
      mv[r][i] = mArr[sb + row];
      lv[r][i] = 1.0f / lArr[sb + row];
    }
  }

  size_t qkbase = ((size_t)(b * Hx + h)) * Rx * Nx * DHx;
  const bf16_t* Vt = vt + ((size_t)(b * Hx + h)) * DHx * Nx;

  // Q fragments are invariant over the column loop: hoist (4 routes x 2 k-steps).
  v16bf aq[Rx][2];
#pragma unroll
  for (int r = 0; r < Rx; r++) {
    const bf16_t* Q = qb + qkbase + (size_t)r * Nx * DHx + (size_t)rowTile * 16 * DHx;
    aq[r][0] = load_a_frag(Q, DHx, lane);
    aq[r][1] = load_a_frag(Q + 32, DHx, lane);
  }

  v8f o[4];
#pragma unroll
  for (int ts = 0; ts < 4; ts++) o[ts] = (v8f){};

  for (int pair = 0; pair < Nx / 32; pair++) {
#pragma unroll
    for (int sub = 0; sub < 2; sub++) {
      int ctile = pair * 2 + sub;
      float pm[8];
#pragma unroll
      for (int r = 0; r < Rx; r++) {
        const bf16_t* Kp = kb + qkbase + (size_t)r * Nx * DHx + (size_t)ctile * 16 * DHx;
        v8f s = {};
        s = wmma_bf16(aq[r][0], load_b_frag(Kp, DHx, lane), s);
        s = wmma_bf16(aq[r][1], load_b_frag(Kp + 32, DHx, lane), s);
#pragma unroll
        for (int i = 0; i < 8; i++) {
          float p = __expf(s[i] - mv[r][i]) * lv[r][i];
          pm[i] = (r == 0) ? p : fmaxf(pm[i], p);
        }
      }
#pragma unroll
      for (int i = 0; i < 8; i++)
        lds[wslot][i + 8 * half][sub * 16 + (lane & 15)] = (bf16_t)pm[i];
    }
    // same-wave DS ops are in-order; make sure stores have landed & block reordering
    asm volatile("s_wait_dscnt 0" ::: "memory");

    v16bf aP;
    {
      int row = lane & 15, koff = half * 8;
      v8bf lo = *(const v8bf*)&lds[wslot][row][koff];
      v8bf hi = *(const v8bf*)&lds[wslot][row][koff + 16];
      aP = cat8(lo, hi);
    }
    int seqBase = pair * 32;
#pragma unroll
    for (int ts = 0; ts < 4; ts++) {
      const bf16_t* Bp = Vt + (size_t)(ts * 16) * Nx + seqBase;  // [k][n] at Bp[n*Nx + k]
      o[ts] = wmma_bf16(aP, load_b_frag(Bp, Nx, lane), o[ts]);
    }
    asm volatile("" ::: "memory");  // keep next-iter LDS stores below this iter's LDS loads
  }

#pragma unroll
  for (int ts = 0; ts < 4; ts++)
#pragma unroll
    for (int i = 0; i < 8; i++) {
      int row = rowTile * 16 + i + 8 * half;
      int col = h * DHx + ts * 16 + (lane & 15);
      ao[((size_t)b * Nx + row) * INNERx + col] = (bf16_t)(o[ts][i]);
    }
}

// ---------------- output projection: out = ao @ W_out^T + b_out (64x32 per wave) ----------------
__global__ void __attribute__((amdgpu_waves_per_eu(1)))
k_outproj(const bf16_t* __restrict__ ao, const bf16_t* __restrict__ wo,
          const float* __restrict__ bout, float* __restrict__ out) {
  int wid  = blockIdx.x * (blockDim.x / WAVE) + (int)(threadIdx.x / WAVE);
  int lane = threadIdx.x & (WAVE - 1);
  const int CB = DIMx / 32;  // 24
  if (wid >= (Bx * Nx / 64) * CB) return;
  int cblk = wid % CB, rowBlk = wid / CB;
  int mBase = rowBlk * 64, cBase = cblk * 32;

  const bf16_t* A0 = ao + (size_t)mBase * INNERx;
  const bf16_t* B0 = wo + (size_t)cBase * INNERx;

  v8f acc[4][2];
#pragma unroll
  for (int t = 0; t < 4; t++)
#pragma unroll
    for (int u = 0; u < 2; u++) acc[t][u] = (v8f){};

  for (int k0 = 0; k0 < INNERx; k0 += 32) {
    __builtin_prefetch(A0 + k0 + 128, 0, 1);
    __builtin_prefetch(B0 + k0 + 128, 0, 1);
    v16bf af[4], bf[2];
#pragma unroll
    for (int t = 0; t < 4; t++) af[t] = load_a_frag(A0 + (size_t)t * 16 * INNERx + k0, INNERx, lane);
#pragma unroll
    for (int u = 0; u < 2; u++) bf[u] = load_b_frag(B0 + (size_t)u * 16 * INNERx + k0, INNERx, lane);
#pragma unroll
    for (int t = 0; t < 4; t++)
#pragma unroll
      for (int u = 0; u < 2; u++) acc[t][u] = wmma_bf16(af[t], bf[u], acc[t][u]);
  }

  int half = lane >> 4;
#pragma unroll
  for (int t = 0; t < 4; t++)
#pragma unroll
    for (int u = 0; u < 2; u++) {
      int col = cBase + u * 16 + (lane & 15);
      float bias = bout[col];
#pragma unroll
      for (int i = 0; i < 8; i++) {
        int m = mBase + t * 16 + i + 8 * half;
        out[(size_t)m * DIMx + col] = acc[t][u][i] + bias;
      }
    }
}

extern "C" void kernel_launch(void* const* d_in, const int* in_sizes, int n_in,
                              void* d_out, int out_size, void* d_ws, size_t ws_size,
                              hipStream_t stream) {
  const float* x     = (const float*)d_in[0];   // (2,1024,768)
  const float* W_qkv = (const float*)d_in[1];   // (4,2304,768)
  const float* W_out = (const float*)d_in[2];   // (768,768)
  const float* b_out = (const float*)d_in[3];   // (768,)
  float* out = (float*)d_out;                   // (2,1024,768)
  (void)in_sizes; (void)n_in; (void)out_size; (void)ws_size;

  char* ws = (char*)d_ws;
  auto carve = [&](size_t bytes) {
    void* p = (void*)ws;
    ws += (bytes + 255) & ~(size_t)255;
    return p;
  };
  const size_t nX  = (size_t)Bx * Nx * DIMx;           // 1,572,864
  const size_t nWq = (size_t)Rx * 3 * INNERx * DIMx;   // 7,077,888
  const size_t nWo = (size_t)DIMx * INNERx;            //   589,824
  const size_t nQK = (size_t)Bx * Hx * Rx * Nx * DHx;  // 6,291,456
  const size_t nV  = (size_t)Bx * Hx * Nx * DHx;       // 1,572,864
  const size_t nSt = (size_t)Bx * Hx * Rx * Nx;        //    98,304

  bf16_t* xb  = (bf16_t*)carve(nX  * 2);
  bf16_t* wqb = (bf16_t*)carve(nWq * 2);
  bf16_t* wob = (bf16_t*)carve(nWo * 2);
  bf16_t* qb  = (bf16_t*)carve(nQK * 2);
  bf16_t* kb  = (bf16_t*)carve(nQK * 2);
  bf16_t* vt  = (bf16_t*)carve(nV  * 2);
  float*  mA  = (float*) carve(nSt * 4);
  float*  lA  = (float*) carve(nSt * 4);
  bf16_t* ao  = (bf16_t*)carve(nX  * 2);

  // fp32 -> bf16 conversions
  k_cvt_bf16<<<(int)((nX  + 255) / 256), 256, 0, stream>>>(x,     xb,  (int)nX);
  k_cvt_bf16<<<(int)((nWq + 255) / 256), 256, 0, stream>>>(W_qkv, wqb, (int)nWq);
  k_cvt_bf16<<<(int)((nWo + 255) / 256), 256, 0, stream>>>(W_out, wob, (int)nWo);

  // QKV projection: 32 row-blocks x 216 (route,col)-blocks, 8 waves/block
  k_qkv<<<32 * 216 / 8, 256, 0, stream>>>(xb, wqb, qb, kb, vt);

  // attention pass 1 (stats): B*H*R*16 = 1536 waves
  k_attn_stats<<<1536 / 8, 256, 0, stream>>>(qb, kb, mA, lA);

  // attention pass 2 (apply): B*H*64 = 1536 waves
  k_attn_apply<<<1536 / 8, 256, 0, stream>>>(qb, kb, vt, mA, lA, ao);

  // output projection: 32 row-blocks x 24 col-blocks
  k_outproj<<<32 * 24 / 8, 256, 0, stream>>>(ao, wob, b_out, out);
}
